// Attention_19954418057611
// MI455X (gfx1250) — matrix operand order repaired and checked by reading, not compiled, not run on hardware
//
#include <hip/hip_runtime.h>
#include <stdint.h>

typedef _Float16 f16;
typedef __attribute__((ext_vector_type(16))) _Float16 v16h;
typedef __attribute__((ext_vector_type(8)))  float    v8f;

#define WMMA_F16(a, b, c) __builtin_amdgcn_wmma_f32_16x16x32_f16( \
    false, (a), false, (b), (short)0, (c), false, false)

static constexpr int Bc = 2, Sc = 2048, Dc = 4096, Hc = 32, KVc = 8, HDc = 128, NREP = 4;

// ---- CDNA5 async global->LDS copy (ASYNCcnt path, CDNA5 ISA) ----

__device__ __forceinline__ void async_b128(uint32_t lds_off, const f16* g) {
    asm volatile("global_load_async_to_lds_b128 %0, %1, off"
                 :: "v"(lds_off), "v"((uint64_t)(uintptr_t)g)
                 : "memory");
}
__device__ __forceinline__ void wait_async_le4() {
    asm volatile("s_wait_asynccnt 0x4" ::: "memory");
}
__device__ __forceinline__ void wait_async_0() {
    asm volatile("s_wait_asynccnt 0x0" ::: "memory");
}

// ---------------- elementwise helpers ----------------

__global__ void cast_f32_f16(const float* __restrict__ in, f16* __restrict__ out, int n) {
    int i = blockIdx.x * blockDim.x + threadIdx.x;
    if (i < n) out[i] = (f16)in[i];
}

// in: K x N row-major f32 ; out: N x K row-major f16 (B^T so WMMA B-frags read contiguous K)
__global__ void transpose_cast(const float* __restrict__ in, f16* __restrict__ out, int K, int N) {
    int i = blockIdx.x * blockDim.x + threadIdx.x;
    if (i < K * N) {
        int k = i / N, n = i % N;
        out[(size_t)n * K + k] = (f16)in[i];
    }
}

// interleaved-pair RoPE on an f16 activation matrix of shape (B*S) x (nheads*HD)
__global__ void rope_kernel(f16* __restrict__ T, const float* __restrict__ cosT,
                            const float* __restrict__ sinT, int nheads, float scale) {
    const int HP = HDc / 2;
    int i = blockIdx.x * blockDim.x + threadIdx.x;
    int total = Bc * Sc * nheads * HP;
    if (i >= total) return;
    int p = i % HP;
    int h = (i / HP) % nheads;
    int s = (i / (HP * nheads)) % Sc;
    int b = i / (HP * nheads * Sc);
    int ld = nheads * HDc;
    size_t off = (size_t)(b * Sc + s) * ld + h * HDc + 2 * p;
    float t0 = (float)T[off], t1 = (float)T[off + 1];
    float c = cosT[s * HP + p], sn = sinT[s * HP + p];
    T[off]     = (f16)((t0 * c - t1 * sn) * scale);
    T[off + 1] = (f16)((t0 * sn + t1 * c) * scale);
}

// ---------------- LDS-staged, async double-buffered WMMA GEMM ----------------
// C(MxN) = A(MxK) * B(KxN), B passed as BT(NxK).
// 256 threads = 8 waves; block tile 128x128, K-tile 32.
// Wave (wm = wave&3, wn = wave>>2) computes a 32x64 strip: 2x4 WMMA tiles.

template <typename OUTT>
__global__ __launch_bounds__(256) void gemm_lds(const f16* __restrict__ A,
                                                const f16* __restrict__ BT,
                                                OUTT* __restrict__ C,
                                                int M, int N, int Kd) {
    __shared__ f16 Ab[2][128][40];   // rows padded 32->40 f16 to spread LDS banks
    __shared__ f16 Bb[2][128][40];
    const int tid = threadIdx.x;
    const int lane = tid & 31, wave = tid >> 5;
    const int kg = lane >> 4, lm = lane & 15;
    const int m0 = blockIdx.y * 128, n0 = blockIdx.x * 128;
    const int wm = wave & 3, wn = wave >> 2;

    // async-copy assignment: 512 x 16B chunks per matrix, 2 each per thread
    const int crow = tid >> 2, cpart = tid & 3;

    auto issue = [&](int buf, int k0) {
        const f16* ga0 = A  + (size_t)(m0 + crow)      * Kd + k0 + cpart * 8;
        const f16* ga1 = A  + (size_t)(m0 + 64 + crow) * Kd + k0 + cpart * 8;
        const f16* gb0 = BT + (size_t)(n0 + crow)      * Kd + k0 + cpart * 8;
        const f16* gb1 = BT + (size_t)(n0 + 64 + crow) * Kd + k0 + cpart * 8;
        async_b128((uint32_t)(uintptr_t)&Ab[buf][crow][cpart * 8],      ga0);
        async_b128((uint32_t)(uintptr_t)&Ab[buf][64 + crow][cpart * 8], ga1);
        async_b128((uint32_t)(uintptr_t)&Bb[buf][crow][cpart * 8],      gb0);
        async_b128((uint32_t)(uintptr_t)&Bb[buf][64 + crow][cpart * 8], gb1);
    };

    v8f acc[2][4] = {};
    const int KT = Kd >> 5;

    issue(0, 0);
    for (int kt = 0; kt < KT; kt++) {
        const int cur = kt & 1;
        if (kt + 1 < KT) {
            issue(cur ^ 1, (kt + 1) << 5);   // prefetch next tile into other buffer
            wait_async_le4();                // oldest 4 (current tile) complete, in order
        } else {
            wait_async_0();
        }
        __syncthreads();

        v16h afrag[2];
#pragma unroll
        for (int tm = 0; tm < 2; tm++) {
            const f16* ap = &Ab[cur][wm * 32 + tm * 16 + lm][kg * 8];
#pragma unroll
            for (int i = 0; i < 8; i++) { afrag[tm][i] = ap[i]; afrag[tm][8 + i] = ap[16 + i]; }
        }
#pragma unroll
        for (int tn = 0; tn < 4; tn++) {
            v16h b;
            const f16* bp = &Bb[cur][wn * 64 + tn * 16 + lm][kg * 8];
#pragma unroll
            for (int i = 0; i < 8; i++) { b[i] = bp[i]; b[8 + i] = bp[16 + i]; }
            acc[0][tn] = WMMA_F16(afrag[0], b, acc[0][tn]);
            acc[1][tn] = WMMA_F16(afrag[1], b, acc[1][tn]);
        }
        __syncthreads();                     // all waves done with buf[cur] before reuse
    }

#pragma unroll
    for (int tm = 0; tm < 2; tm++)
#pragma unroll
        for (int tn = 0; tn < 4; tn++)
#pragma unroll
            for (int v = 0; v < 8; v++) {
                int row = m0 + wm * 32 + tm * 16 + kg * 8 + v;
                int col = n0 + wn * 64 + tn * 16 + lm;
                C[(size_t)row * N + col] = (OUTT)acc[tm][tn][v];
            }
}

// ---------------- attention: one workgroup per (b, h, 16-query tile) ----------------
// 8 waves. LDS holds the full 16 x S f32 score/prob tile (128 KB of the WGP's 320 KB).

__global__ void attn_kernel(const f16* __restrict__ Q, const f16* __restrict__ Km,
                            const f16* __restrict__ Vm, f16* __restrict__ O) {
    extern __shared__ float sc[];   // 16 * Sc floats
    const int lane = threadIdx.x & 31, wave = threadIdx.x >> 5;
    const int kg = lane >> 4, lm = lane & 15;
    const int bid = blockIdx.x;
    const int qt = bid % (Sc / 16);
    const int h  = (bid / (Sc / 16)) % Hc;
    const int b  = bid / ((Sc / 16) * Hc);
    const int kvh = h / NREP;
    const int qbase = qt * 16;
    const int ldq = Hc * HDc, ldk = KVc * HDc;

    const f16* qrow = Q + (size_t)(b * Sc + qbase + lm) * ldq + h * HDc;

    // Phase 1: scores = (Q/sqrt(HD)) @ K^T into LDS, causal-masked.
    for (int t = 0; t < 16; t++) {
        int jb = (t * 8 + wave) * 16;
        v8f acc = {};
        for (int d0 = 0; d0 < HDc; d0 += 32) {
            v16h a, bv;
            const f16* ap = qrow + d0 + kg * 8;
#pragma unroll
            for (int i = 0; i < 8; i++) { a[i] = ap[i]; a[8 + i] = ap[16 + i]; }
            const f16* kp = Km + (size_t)(b * Sc + jb + lm) * ldk + kvh * HDc + d0 + kg * 8;
#pragma unroll
            for (int i = 0; i < 8; i++) { bv[i] = kp[i]; bv[8 + i] = kp[16 + i]; }
            acc = WMMA_F16(a, bv, acc);
        }
#pragma unroll
        for (int v = 0; v < 8; v++) {
            int m = kg * 8 + v;
            int key = jb + lm;
            float val = acc[v];
            if (key > qbase + m) val = -3.0e38f;   // causal mask
            sc[m * Sc + key] = val;
        }
    }
    __syncthreads();

    // Phase 2: row softmax (16 rows, one thread each)
    if (threadIdx.x < 16) {
        float* row = sc + threadIdx.x * Sc;
        float mx = -3.0e38f;
        for (int j = 0; j < Sc; j++) mx = fmaxf(mx, row[j]);
        float sum = 0.f;
        for (int j = 0; j < Sc; j++) { float e = __expf(row[j] - mx); row[j] = e; sum += e; }
        float inv = 1.f / sum;
        for (int j = 0; j < Sc; j++) row[j] *= inv;
    }
    __syncthreads();

    // Phase 3: out = P @ V ; wave w owns head-dim columns [w*16, w*16+16)
    const int n0 = wave * 16;
    v8f acc = {};
    for (int k0 = 0; k0 < Sc; k0 += 32) {
        v16h a, bv;
        const float* pr = sc + lm * Sc + k0 + kg * 8;
#pragma unroll
        for (int i = 0; i < 8; i++) { a[i] = (f16)pr[i]; a[8 + i] = (f16)pr[16 + i]; }
        const f16* vp = Vm + (size_t)(b * Sc + k0 + kg * 8) * ldk + kvh * HDc + n0 + lm;
#pragma unroll
        for (int i = 0; i < 8; i++) { bv[i] = vp[(size_t)i * ldk]; bv[8 + i] = vp[(size_t)(16 + i) * ldk]; }
        acc = WMMA_F16(a, bv, acc);
    }
#pragma unroll
    for (int v = 0; v < 8; v++) {
        int row = b * Sc + qbase + kg * 8 + v;
        O[(size_t)row * ldq + h * HDc + n0 + lm] = (f16)acc[v];
    }
}

// ---------------- launch ----------------

extern "C" void kernel_launch(void* const* d_in, const int* in_sizes, int n_in,
                              void* d_out, int out_size, void* d_ws, size_t ws_size,
                              hipStream_t stream) {
    const float* x  = (const float*)d_in[0];
    const float* fc = (const float*)d_in[1];
    const float* fs = (const float*)d_in[2];
    // d_in[3] = mask: unused, causal structure applied analytically
    const float* wq = (const float*)d_in[4];
    const float* wk = (const float*)d_in[5];
    const float* wv = (const float*)d_in[6];
    const float* wo = (const float*)d_in[7];

    char* ws = (char*)d_ws;
    f16* xh  = (f16*)(ws);
    f16* wqT = (f16*)(ws +  33554432ull);
    f16* wkT = (f16*)(ws +  67108864ull);
    f16* wvT = (f16*)(ws +  75497472ull);
    f16* woT = (f16*)(ws +  83886080ull);
    f16* Qh  = (f16*)(ws + 117440512ull);
    f16* Kh  = (f16*)(ws + 150994944ull);
    f16* Vh  = (f16*)(ws + 159383552ull);
    f16* Oh  = (f16*)(ws + 167772160ull);   // total 192 MB

    const int BS = Bc * Sc;                 // 4096
    const int nX = BS * Dc;

    cast_f32_f16<<<(nX + 255) / 256, 256, 0, stream>>>(x, xh, nX);
    transpose_cast<<<(Dc * Hc  * HDc + 255) / 256, 256, 0, stream>>>(wq, wqT, Dc, Hc * HDc);
    transpose_cast<<<(Dc * KVc * HDc + 255) / 256, 256, 0, stream>>>(wk, wkT, Dc, KVc * HDc);
    transpose_cast<<<(Dc * KVc * HDc + 255) / 256, 256, 0, stream>>>(wv, wvT, Dc, KVc * HDc);
    transpose_cast<<<(Hc * HDc * Dc + 255) / 256, 256, 0, stream>>>(wo, woT, Hc * HDc, Dc);

    gemm_lds<f16><<<dim3((Hc  * HDc) / 128, BS / 128), 256, 0, stream>>>(xh, wqT, Qh, BS, Hc  * HDc, Dc);
    gemm_lds<f16><<<dim3((KVc * HDc) / 128, BS / 128), 256, 0, stream>>>(xh, wkT, Kh, BS, KVc * HDc, Dc);
    gemm_lds<f16><<<dim3((KVc * HDc) / 128, BS / 128), 256, 0, stream>>>(xh, wvT, Vh, BS, KVc * HDc, Dc);

    const int nQr = Bc * Sc * Hc  * (HDc / 2);
    const int nKr = Bc * Sc * KVc * (HDc / 2);
    rope_kernel<<<(nQr + 255) / 256, 256, 0, stream>>>(Qh, fc, fs, Hc, 0.08838834764831845f); // 1/sqrt(128)
    rope_kernel<<<(nKr + 255) / 256, 256, 0, stream>>>(Kh, fc, fs, KVc, 1.0f);

    attn_kernel<<<Bc * Hc * (Sc / 16), 256, 16 * Sc * sizeof(float), stream>>>(Qh, Kh, Vh, Oh);

    gemm_lds<float><<<dim3(Dc / 128, BS / 128), 256, 0, stream>>>(Oh, woT, (float*)d_out, BS, Dc, Dc);
}
